// Model_75737453298115
// MI455X (gfx1250) — matrix operand order repaired and checked by reading, not compiled, not run on hardware
//
#include <hip/hip_runtime.h>
#include <stdint.h>

// ---------------- problem constants ----------------
#define N_FILTERS 2048
#define FILTER_LEN 2048
#define K_TOP 2048
#define T_IN  16384
#define PAD   1024
#define T_OUT 16385                         // T_IN + 2*PAD - FILTER_LEN + 1
#define XP_MAX (T_OUT + FILTER_LEN - 1)     // padded-signal coordinate space
#define NTOT (N_FILTERS * T_OUT)            // 33,556,480 (< 2^31)

// ---------------- GEMM tiling ----------------
#define BM 32          // filters per workgroup (2 waves in M)
#define BN 256         // time samples per workgroup (4 waves in N, 64 each)
#define XW_LEN 2304    // x window halves needed: BN + FILTER_LEN - 1 = 2303 -> 2304 (16B-aligned rows)

#define BINS1 8192     // level-1: abs-bits >> 18
#define BINS2 262144   // level-2: abs-bits & 0x3FFFF

typedef __bf16 bf16_t;
typedef bf16_t v16bf __attribute__((ext_vector_type(16)));
typedef float  v8f   __attribute__((ext_vector_type(8)));

__device__ __forceinline__ unsigned short f32_to_bf16_rne(float v) {
    unsigned u = __float_as_uint(v);
    unsigned r = u + 0x7fffu + ((u >> 16) & 1u);   // round-to-nearest-even
    return (unsigned short)(r >> 16);
}

// ---------------------------------------------------------------------------
// K1: L2-normalize atoms -> f32 table (for recon) + bf16 table (for WMMA)
// ---------------------------------------------------------------------------
__global__ void __launch_bounds__(256) prep_atoms(const float* __restrict__ atoms,
                                                  float* __restrict__ aF,
                                                  unsigned short* __restrict__ aH) {
    const int f = blockIdx.x;
    const float* src = atoms + (size_t)f * FILTER_LEN;
    __shared__ float red[256];
    float s = 0.f;
    for (int k = threadIdx.x; k < FILTER_LEN; k += 256) { float v = src[k]; s += v * v; }
    red[threadIdx.x] = s;
    __syncthreads();
    for (int off = 128; off > 0; off >>= 1) {
        if (threadIdx.x < off) red[threadIdx.x] += red[threadIdx.x + off];
        __syncthreads();
    }
    const float inv = 1.f / (sqrtf(red[0]) + 1e-12f);
    for (int k = threadIdx.x; k < FILTER_LEN; k += 256) {
        float v = src[k] * inv;
        aF[(size_t)f * FILTER_LEN + k] = v;
        aH[(size_t)f * FILTER_LEN + k] = f32_to_bf16_rne(v);
    }
}

// ---------------------------------------------------------------------------
// K2: matched-filter GEMM via v_wmma_f32_16x16x32_bf16
//   fm[f,t] = sum_k a[f,k] * xpad[t+k],  xpad[i] = x[i-PAD] (zero padded)
// A = atoms tile (M=filters), B = Hankel window of x (N=time).
// 8 staggered LDS copies of the x window give 16B-aligned ds_load_b128 for B.
// ---------------------------------------------------------------------------
__global__ void __launch_bounds__(256) mf_gemm(const unsigned short* __restrict__ aH,
                                               const float* __restrict__ x,
                                               float* __restrict__ fm) {
    __shared__ __align__(16) unsigned short xs[8][XW_LEN];   // 36 KB

    const int t0 = blockIdx.y * BN;
    const int f0 = blockIdx.x * BM;

    // Stage staggered bf16 x-window: xs[c][j] = xpad[t0 + j + c]
    for (int i = threadIdx.x; i < 8 * XW_LEN; i += 256) {
        const int c = i / XW_LEN, j = i - c * XW_LEN;
        const int xi = t0 + j + c - PAD;
        float v = (xi >= 0 && xi < T_IN) ? x[xi] : 0.f;
        xs[c][j] = f32_to_bf16_rne(v);
    }
    __syncthreads();

    const int tid  = threadIdx.x;
    const int wave = tid >> 5, lane = tid & 31;
    const int wm = wave >> 2, wn = wave & 3;      // 2x4 wave grid
    const int lo = lane & 15, hi = lane >> 4;

    union Frag { uint4 q[2]; v16bf v; };

    v8f acc0 = {0,0,0,0,0,0,0,0};
    v8f acc[4] = {acc0, acc0, acc0, acc0};

    // A row base: 16B-aligned (2*(row*2048 + hi*16 + kk) with kk % 32 == 0)
    const unsigned short* aRow =
        aH + (size_t)(f0 + wm * 16 + lo) * FILTER_LEN + hi * 16;
    const int tb = wn * 64 + lo;                  // column base for this lane

    for (int kk = 0; kk < FILTER_LEN; kk += 32) {
        Frag A;
        A.q[0] = *(const uint4*)(aRow + kk);       // k = kk + hi*16 + [0..7]
        A.q[1] = *(const uint4*)(aRow + kk + 8);   // k = kk + hi*16 + [8..15]
#pragma unroll
        for (int j = 0; j < 4; ++j) {
            const int o = tb + j * 16 + kk + hi * 16;  // half-offset of B[k..k+15, n]
            const int c = o & 7;
            const unsigned short* bp = &xs[c][o - c];  // 16B-aligned
            Frag B;
            B.q[0] = *(const uint4*)(bp);
            B.q[1] = *(const uint4*)(bp + 8);
            acc[j] = __builtin_amdgcn_wmma_f32_16x16x32_bf16(
                false, A.v, false, B.v, (short)0, acc[j], false, false);
        }
    }

    // C/D layout: VGPR r -> M = r + hi*8, N = lo
#pragma unroll
    for (int j = 0; j < 4; ++j) {
        const int t = t0 + wn * 64 + j * 16 + lo;
        if (t < T_OUT) {
            float* dst = fm + (size_t)(f0 + wm * 16 + hi * 8) * T_OUT + t;
#pragma unroll
            for (int r = 0; r < 8; ++r) dst[(size_t)r * T_OUT] = acc[j][r];
        }
    }
}

// ---------------------------------------------------------------------------
// K3a: level-1 histogram of abs-bits >> 18 (LDS-local, then merged)
// ---------------------------------------------------------------------------
__global__ void __launch_bounds__(256) hist1(const float* __restrict__ fm,
                                             unsigned* __restrict__ gh) {
    __shared__ unsigned lh[BINS1];                // 32 KB
    for (int b = threadIdx.x; b < BINS1; b += 256) lh[b] = 0;
    __syncthreads();
    const int stride = gridDim.x * 256;
    for (int i = blockIdx.x * 256 + threadIdx.x; i < NTOT; i += stride) {
        unsigned u = __float_as_uint(fm[i]) & 0x7fffffffu;
        atomicAdd(&lh[u >> 18], 1u);
    }
    __syncthreads();
    for (int b = threadIdx.x; b < BINS1; b += 256)
        if (lh[b]) atomicAdd(&gh[b], lh[b]);
}

// K3b: scan level-1 from the top to find threshold bin + residual count
__global__ void __launch_bounds__(256) scan1(const unsigned* __restrict__ gh,
                                             int* __restrict__ scal) {
    __shared__ unsigned part[256];
    const int q = threadIdx.x;
    unsigned s = 0;
    const int hib = BINS1 - 32 * q;
    for (int b = hib - 32; b < hib; ++b) s += gh[b];
    part[q] = s;
    __syncthreads();
    if (threadIdx.x == 0) {
        unsigned cum = 0; int Bstar = 0; unsigned need = K_TOP;
        for (int qq = 0; qq < 256; ++qq) {
            if (cum + part[qq] >= (unsigned)K_TOP) {
                const int hb = BINS1 - 32 * qq;
                for (int b = hb - 1; b >= hb - 32; --b) {
                    unsigned c = gh[b];
                    if (cum + c >= (unsigned)K_TOP) { Bstar = b; need = K_TOP - cum; break; }
                    cum += c;
                }
                break;
            }
            cum += part[qq];
        }
        scal[0] = Bstar;
        scal[1] = (int)need;
    }
}

// K3c: level-2 histogram of low 18 abs-bits, restricted to the threshold bin
__global__ void __launch_bounds__(256) hist2(const float* __restrict__ fm,
                                             const int* __restrict__ scal,
                                             unsigned* __restrict__ gh2) {
    const unsigned Bstar = (unsigned)scal[0];
    const int stride = gridDim.x * 256;
    for (int i = blockIdx.x * 256 + threadIdx.x; i < NTOT; i += stride) {
        unsigned u = __float_as_uint(fm[i]) & 0x7fffffffu;
        if ((u >> 18) == Bstar) atomicAdd(&gh2[u & 0x3ffffu], 1u);
    }
}

// K3d: scan level-2 -> exact 31-bit threshold W and tie budget R2
__global__ void __launch_bounds__(256) scan2(const unsigned* __restrict__ gh2,
                                             int* __restrict__ scal) {
    __shared__ unsigned part[256];
    const unsigned need = (unsigned)scal[1];
    const int q = threadIdx.x;
    unsigned s = 0;
    const int hib = BINS2 - 1024 * q;
    for (int b = hib - 1024; b < hib; ++b) s += gh2[b];
    part[q] = s;
    __syncthreads();
    if (threadIdx.x == 0) {
        unsigned cum = 0, sstar = 0, r2 = need;
        for (int qq = 0; qq < 256; ++qq) {
            if (cum + part[qq] >= need) {
                const int hb = BINS2 - 1024 * qq;
                for (int b = hb - 1; b >= hb - 1024; --b) {
                    unsigned c = gh2[b];
                    if (cum + c >= need) { sstar = (unsigned)b; r2 = need - cum; break; }
                    cum += c;
                }
                break;
            }
            cum += part[qq];
        }
        scal[2] = (int)(((unsigned)scal[0] << 18) | sstar);  // threshold bit-pattern
        scal[3] = (int)r2;                                   // ties to accept at == W
    }
}

// ---------------------------------------------------------------------------
// K4: sparsify fm in place (scatter ABS values, zero elsewhere) + compact nz list
// ---------------------------------------------------------------------------
__global__ void __launch_bounds__(256) select_sparsify(float* __restrict__ fm,
                                                       const int* __restrict__ scal,
                                                       int* __restrict__ ctr,
                                                       int* __restrict__ nzF,
                                                       int* __restrict__ nzT,
                                                       float* __restrict__ nzV) {
    const unsigned W = (unsigned)scal[2];
    const int R2 = scal[3];
    const int stride = gridDim.x * 256;
    for (int i = blockIdx.x * 256 + threadIdx.x; i < NTOT; i += stride) {
        const float v = fm[i];
        const unsigned u = __float_as_uint(v) & 0x7fffffffu;
        bool keep = u > W;
        if (u == W) keep = (atomicAdd(&ctr[0], 1) < R2);
        const float av = __uint_as_float(u);     // |v|
        if (keep) {
            const int slot = atomicAdd(&ctr[1], 1);
            if (slot < K_TOP) {
                const int f = i / T_OUT;
                nzF[slot] = f;
                nzT[slot] = i - f * T_OUT;
                nzV[slot] = av;
            }
            fm[i] = av;
        } else {
            fm[i] = 0.f;
        }
    }
}

// ---------------------------------------------------------------------------
// K5: transpose-conv reconstruction (gather over the 2048-entry nz list)
//   recon[t] = sum_nz v * a[f, t - tau + PAD]
// ---------------------------------------------------------------------------
__global__ void __launch_bounds__(256) recon_kernel(const float* __restrict__ aF,
                                                    const int* __restrict__ nzF,
                                                    const int* __restrict__ nzT,
                                                    const float* __restrict__ nzV,
                                                    const int* __restrict__ ctr,
                                                    float* __restrict__ recon) {
    __shared__ int   sF[256];
    __shared__ int   sT[256];
    __shared__ float sV[256];
    const int t = blockIdx.x * 256 + threadIdx.x;
    const int count = min(ctr[1], K_TOP);
    float acc = 0.f;
    for (int base = 0; base < count; base += 256) {
        const int n = min(256, count - base);
        __syncthreads();
        if (threadIdx.x < n) {
            sF[threadIdx.x] = nzF[base + threadIdx.x];
            sT[threadIdx.x] = nzT[base + threadIdx.x];
            sV[threadIdx.x] = nzV[base + threadIdx.x];
        }
        __syncthreads();
        for (int j = 0; j < n; ++j) {
            const int k = t - sT[j] + PAD;
            if ((unsigned)k < (unsigned)FILTER_LEN)
                acc += sV[j] * aF[(size_t)sF[j] * FILTER_LEN + k];
        }
    }
    if (t < T_IN) recon[t] = acc;
}

// ---------------------------------------------------------------------------
extern "C" void kernel_launch(void* const* d_in, const int* in_sizes, int n_in,
                              void* d_out, int out_size, void* d_ws, size_t ws_size,
                              hipStream_t stream) {
    const float* x     = (const float*)d_in[0];   // [1,1,16384]
    const float* atoms = (const float*)d_in[1];   // [2048,1,2048]

    char* ws = (char*)d_ws;
    size_t off = 0;
    float* aF = (float*)(ws + off);           off += (size_t)N_FILTERS * FILTER_LEN * 4; // 16 MB
    unsigned short* aH = (unsigned short*)(ws + off);
                                              off += (size_t)N_FILTERS * FILTER_LEN * 2; // 8 MB
    const size_t zero_from = off;
    unsigned* gh1 = (unsigned*)(ws + off);    off += (size_t)BINS1 * 4;                  // 32 KB
    unsigned* gh2 = (unsigned*)(ws + off);    off += (size_t)BINS2 * 4;                  // 1 MB
    int* scal = (int*)(ws + off);             off += 256;  // [0]=Bstar [1]=need [2]=W [3]=R2 [4]=tie [5]=nzCount
    int*   nzF = (int*)(ws + off);            off += (size_t)K_TOP * 4;
    int*   nzT = (int*)(ws + off);            off += (size_t)K_TOP * 4;
    float* nzV = (float*)(ws + off);          off += (size_t)K_TOP * 4;
    const size_t zero_bytes = off - zero_from;

    float* fm    = (float*)d_out;                       // sparse output region [2048 x 16385]
    float* recon = fm + (size_t)N_FILTERS * T_OUT;      // recon region [16384]

    hipMemsetAsync(ws + zero_from, 0, zero_bytes, stream);

    prep_atoms<<<N_FILTERS, 256, 0, stream>>>(atoms, aF, aH);

    dim3 gg(N_FILTERS / BM, (T_OUT + BN - 1) / BN);     // 64 x 65
    mf_gemm<<<gg, 256, 0, stream>>>(aH, x, fm);

    hist1<<<2048, 256, 0, stream>>>(fm, gh1);
    scan1<<<1, 256, 0, stream>>>(gh1, scal);
    hist2<<<2048, 256, 0, stream>>>(fm, scal, gh2);
    scan2<<<1, 256, 0, stream>>>(gh2, scal);
    select_sparsify<<<2048, 256, 0, stream>>>(fm, scal, scal + 4, nzF, nzT, nzV);
    recon_kernel<<<T_IN / 256, 256, 0, stream>>>(aF, nzF, nzT, nzV, scal + 4, recon);
}
